// myGCN_53455162966062
// MI455X (gfx1250) — compile-verified, matrix operand order checked by reading, not run on hardware
//
#include <hip/hip_runtime.h>
#include <math.h>

#define N_NODES    100000
#define N_EDGES    3200000
#define NUM_GRAPHS 2048
#define IN_CH      35
#define HID        128
#define HALF_HID   64
#define EPS        1e-5f

typedef __attribute__((ext_vector_type(16))) __bf16 v16bf;
typedef __attribute__((ext_vector_type(8)))  __bf16 v8bf;
typedef __attribute__((ext_vector_type(8)))  float  v8f;

union frag16 { v16bf v; v8bf h[2]; };

// ---------------------------------------------------------------- utilities
__global__ void fill_kernel(float* __restrict__ p, long long n, float v) {
    long long i = (long long)blockIdx.x * blockDim.x + threadIdx.x;
    if (i < n) p[i] = v;
}

__global__ void count_deg_kernel(const int* __restrict__ ei, float* __restrict__ deg) {
    long long e = (long long)blockIdx.x * blockDim.x + threadIdx.x;
    if (e < N_EDGES) atomicAdd(&deg[ei[N_EDGES + e]], 1.0f);
}

__global__ void make_dinv_kernel(float* __restrict__ deg) {
    long long i = (long long)blockIdx.x * blockDim.x + threadIdx.x;
    if (i < N_NODES) deg[i] = rsqrtf(deg[i]);   // deg >= 1 always (self loop)
}

// x [N,35] f32 -> actbf [N,64] bf16, zero-padded
__global__ void xpad_kernel(const float* __restrict__ x, __bf16* __restrict__ a) {
    long long idx = (long long)blockIdx.x * blockDim.x + threadIdx.x;
    if (idx >= (long long)N_NODES * 64) return;
    int i = (int)(idx >> 6), k = (int)(idx & 63);
    a[idx] = (k < IN_CH) ? (__bf16)x[(size_t)i * IN_CH + k] : (__bf16)0.0f;
}

// W [K,128] f32 -> fragment-ordered packed bf16.
// Pack index = (((ct*NKS + ks)*32 + lane)*16 + i); lane: h=lane>>4, n=ct*16+(lane&15);
// k = ks*32 + (i/8)*16 + h*8 + (i%8). Zero-pad k >= K.
__global__ void pack_w_kernel(const float* __restrict__ W, __bf16* __restrict__ Wp,
                              int K, int nks) {
    int idx = blockIdx.x * blockDim.x + threadIdx.x;
    int total = nks * 32 * 16 * 8;              // = Kp * 128
    if (idx >= total) return;
    int i    = idx & 15;
    int lane = (idx >> 4) & 31;
    int t    = idx >> 9;
    int ks   = t % nks;
    int ct   = t / nks;
    int h = lane >> 4;
    int n = ct * 16 + (lane & 15);
    int k = ks * 32 + ((i >> 3) << 4) + (h << 3) + (i & 7);
    Wp[idx] = (k < K) ? (__bf16)W[(size_t)k * HID + n] : (__bf16)0.0f;
}

// ---------------------------------------------------------------- WMMA GEMM
// C[N,128] = A_bf16[N,Kp] @ W_packed.  One wave per 16-row strip; the wave's
// A fragments (NKS k-steps) are loaded once as contiguous b128s and reused
// across all 8 column tiles; B fragments are contiguous 32B packed loads.
template <int NKS>
__global__ __launch_bounds__(32)
void gemm_wmma_packed(const __bf16* __restrict__ A, const __bf16* __restrict__ Wp,
                      float* __restrict__ C) {
    const int lane = threadIdx.x;
    const int half = lane >> 4;
    const int mn   = lane & 15;
    const int row0 = blockIdx.x * 16;
    const int Kp   = NKS * 32;
    const __bf16* arow = A + (size_t)(row0 + mn) * Kp;

    frag16 a[NKS];
#pragma unroll
    for (int ks = 0; ks < NKS; ++ks) {
        const __bf16* p = arow + ks * 32 + (half << 3);
        a[ks].h[0] = *reinterpret_cast<const v8bf*>(p);        // K = k0 + h*8 + 0..7
        a[ks].h[1] = *reinterpret_cast<const v8bf*>(p + 16);   // K = k0+16 + h*8 + 0..7
    }
#pragma unroll
    for (int ct = 0; ct < 8; ++ct) {
        v8f acc = {};
#pragma unroll
        for (int ks = 0; ks < NKS; ++ks) {
            v16bf b = *reinterpret_cast<const v16bf*>(
                Wp + (((ct * NKS + ks) * 32 + lane) << 4));
            acc = __builtin_amdgcn_wmma_f32_16x16x32_bf16(false, a[ks].v, false, b,
                                                          (short)0, acc, false, false);
        }
        float* crow = C + (size_t)(row0 + (half << 3)) * HID + ct * 16 + mn;
#pragma unroll
        for (int r = 0; r < 8; ++r)
            crow[(size_t)r * HID] = acc[r];
    }
}

// ---------------------------------------------------------------- GCN pieces
// acc = dinv[i]^2 * hG + bias   (self-loop term initializes accumulator)
__global__ void init_out_kernel(const float* __restrict__ hG,
                                const float* __restrict__ dinv,
                                const float* __restrict__ bias,
                                float* __restrict__ acc) {
    long long idx = (long long)blockIdx.x * blockDim.x + threadIdx.x;
    if (idx >= (long long)N_NODES * HID) return;
    int i = (int)(idx >> 7), c = (int)(idx & 127);
    float di = dinv[i];
    acc[idx] = di * di * hG[idx] + bias[c];
}

// 32 lanes per edge, 4 channels/lane: acc[dst] += dinv[s]*dinv[d] * hG[src]
__global__ __launch_bounds__(256)
void edge_scatter_kernel(const float* __restrict__ hG, float* __restrict__ acc,
                         const int* __restrict__ ei, const float* __restrict__ dinv) {
    long long gid = (long long)blockIdx.x * blockDim.x + threadIdx.x;
    long long e = gid >> 5;
    if (e >= N_EDGES) return;
    int c0 = (int)(gid & 31) << 2;
    int s = ei[e];
    int d = ei[N_EDGES + e];
    float w = dinv[s] * dinv[d];
    const float4 v = *reinterpret_cast<const float4*>(hG + (size_t)s * HID + c0);
    float* a = acc + (size_t)d * HID + c0;
    atomicAdd(a + 0, w * v.x);
    atomicAdd(a + 1, w * v.y);
    atomicAdd(a + 2, w * v.z);
    atomicAdd(a + 3, w * v.w);
}

// ---------------------------------------------------------------- BatchNorm
__global__ __launch_bounds__(HID)
void bn_stats_kernel(const float* __restrict__ h, float* __restrict__ stats) {
    int c = threadIdx.x;
    float s = 0.0f, s2 = 0.0f;
    for (int i = blockIdx.x; i < N_NODES; i += gridDim.x) {
        float v = h[(size_t)i * HID + c];
        s += v; s2 += v * v;
    }
    atomicAdd(&stats[c], s);
    atomicAdd(&stats[HID + c], s2);
}

__global__ void bn_finalize_kernel(float* __restrict__ stats,
                                   const float* __restrict__ g,
                                   const float* __restrict__ be) {
    int c = threadIdx.x;
    float mu  = stats[c] * (1.0f / (float)N_NODES);
    float var = stats[HID + c] * (1.0f / (float)N_NODES) - mu * mu;
    float sc  = g[c] * rsqrtf(var + EPS);
    stats[c]       = sc;
    stats[HID + c] = be[c] - mu * sc;
}

// out = gelu_exact(acc*scale + shift); also emit bf16 copy for next GEMM
__global__ void bn_gelu_kernel(const float* __restrict__ acc,
                               const float* __restrict__ stats,
                               float* __restrict__ out,
                               __bf16* __restrict__ outbf) {
    long long idx = (long long)blockIdx.x * blockDim.x + threadIdx.x;
    if (idx >= (long long)N_NODES * HID) return;
    int c = (int)(idx & 127);
    float y = acc[idx] * stats[c] + stats[HID + c];
    float gv = 0.5f * y * (1.0f + erff(y * 0.70710678118654752f));
    out[idx] = gv;
    outbf[idx] = (__bf16)gv;
}

// ---------------------------------------------------------------- pooling
__global__ void pool_accum_kernel(const float* __restrict__ h,
                                  const int* __restrict__ batch,
                                  float* __restrict__ pooled,
                                  float* __restrict__ cnt) {
    long long idx = (long long)blockIdx.x * blockDim.x + threadIdx.x;
    if (idx >= (long long)N_NODES * HID) return;
    int i = (int)(idx >> 7), c = (int)(idx & 127);
    int b = batch[i];
    atomicAdd(&pooled[(size_t)b * HID + c], h[idx]);
    if (c == 0) atomicAdd(&cnt[b], 1.0f);
}

// ---------------------------------------------------------------- MLP head
__global__ __launch_bounds__(HALF_HID)
void head_kernel(const float* __restrict__ pooled, const float* __restrict__ cnt,
                 const float* __restrict__ Wl1, const float* __restrict__ bl1,
                 const float* __restrict__ Wl2, const float* __restrict__ bl2,
                 float* __restrict__ out) {
    __shared__ float hsum[HALF_HID];
    int g = blockIdx.x, j = threadIdx.x;
    float inv = 1.0f / fmaxf(cnt[g], 1.0f);
    const float* p = pooled + (size_t)g * HID;
    float acc = bl1[j];
    for (int c = 0; c < HID; ++c) acc += (p[c] * inv) * Wl1[c * HALF_HID + j];
    float hv = acc > 0.0f ? acc : expm1f(acc);   // ELU alpha=1
    hsum[j] = hv * Wl2[j];
    __syncthreads();
    for (int s = HALF_HID / 2; s > 0; s >>= 1) {
        if (j < s) hsum[j] += hsum[j + s];
        __syncthreads();
    }
    if (j == 0) out[g] = hsum[0] + bl2[0];
}

// ---------------------------------------------------------------- driver
struct Ws {
    float* dinv; float* hA; float* hB; float* stats; float* pooled; float* cnt;
    __bf16* actbf; __bf16* wpack;
};

static void run_layer(int nks, const float* W, const float* b,
                      const float* g, const float* be,
                      const Ws& w, const int* ei, hipStream_t stream) {
    int kp128 = nks * 32 * HID;
    pack_w_kernel<<<(kp128 + 255) / 256, 256, 0, stream>>>(W, w.wpack, /*K real handled by pad*/ nks * 32, nks);
    // NOTE: for layer 1 the zero-padding of k>=IN_CH is already in actbf; the
    // weight rows k in [IN_CH, 64) are multiplied by zero activations, but we
    // must still zero them in Wp since W only has IN_CH rows. Handled by caller
    // passing real K via pack (see kernel_launch).
    if (nks == 2) gemm_wmma_packed<2><<<N_NODES / 16, 32, 0, stream>>>(w.actbf, w.wpack, w.hA);
    else          gemm_wmma_packed<4><<<N_NODES / 16, 32, 0, stream>>>(w.actbf, w.wpack, w.hA);
    long long nc = (long long)N_NODES * HID;
    int ncb = (int)((nc + 255) / 256);
    init_out_kernel<<<ncb, 256, 0, stream>>>(w.hA, w.dinv, b, w.hB);
    long long et = (long long)N_EDGES * 32;
    edge_scatter_kernel<<<(int)((et + 255) / 256), 256, 0, stream>>>(w.hA, w.hB, ei, w.dinv);
    fill_kernel<<<1, 256, 0, stream>>>(w.stats, 2 * HID, 0.0f);
    bn_stats_kernel<<<256, HID, 0, stream>>>(w.hB, w.stats);
    bn_finalize_kernel<<<1, HID, 0, stream>>>(w.stats, g, be);
    bn_gelu_kernel<<<ncb, 256, 0, stream>>>(w.hB, w.stats, w.hA, w.actbf);
}

extern "C" void kernel_launch(void* const* d_in, const int* in_sizes, int n_in,
                              void* d_out, int out_size, void* d_ws, size_t ws_size,
                              hipStream_t stream) {
    const float* x     = (const float*)d_in[0];
    const int*   ei    = (const int*)  d_in[1];
    const int*   batch = (const int*)  d_in[2];
    const float* W1 = (const float*)d_in[3],  *b1 = (const float*)d_in[4];
    const float* g1 = (const float*)d_in[5],  *be1 = (const float*)d_in[6];
    const float* W2 = (const float*)d_in[7],  *b2 = (const float*)d_in[8];
    const float* g2 = (const float*)d_in[9],  *be2 = (const float*)d_in[10];
    const float* W3 = (const float*)d_in[11], *b3 = (const float*)d_in[12];
    const float* g3 = (const float*)d_in[13], *be3 = (const float*)d_in[14];
    const float* Wl1 = (const float*)d_in[15], *bl1 = (const float*)d_in[16];
    const float* Wl2 = (const float*)d_in[17], *bl2 = (const float*)d_in[18];
    float* out = (float*)d_out;

    float* ws = (float*)d_ws;
    Ws w;
    w.dinv   = ws;                                   // 100000
    w.hA     = ws + 100000;                          // 12,800,000
    w.hB     = w.hA + (size_t)N_NODES * HID;         // 12,800,000
    w.stats  = w.hB + (size_t)N_NODES * HID;         // 256
    w.pooled = w.stats + 2 * HID;                    // 2048*128
    w.cnt    = w.pooled + (size_t)NUM_GRAPHS * HID;  // 2048
    w.actbf  = (__bf16*)(w.cnt + NUM_GRAPHS);        // N*128 bf16
    w.wpack  = w.actbf + (size_t)N_NODES * HID;      // 128*128 bf16

    // degree / dinv
    fill_kernel<<<(N_NODES + 255) / 256, 256, 0, stream>>>(w.dinv, N_NODES, 1.0f);
    count_deg_kernel<<<(N_EDGES + 255) / 256, 256, 0, stream>>>(ei, w.dinv);
    make_dinv_kernel<<<(N_NODES + 255) / 256, 256, 0, stream>>>(w.dinv);

    // ---- layer 1 (K = 35 padded to 64, NKS = 2)
    xpad_kernel<<<(int)(((long long)N_NODES * 64 + 255) / 256), 256, 0, stream>>>(x, w.actbf);
    {
        int nks = 2, kp128 = nks * 32 * HID;
        pack_w_kernel<<<(kp128 + 255) / 256, 256, 0, stream>>>(W1, w.wpack, IN_CH, nks);
        gemm_wmma_packed<2><<<N_NODES / 16, 32, 0, stream>>>(w.actbf, w.wpack, w.hA);
        long long nc = (long long)N_NODES * HID;
        int ncb = (int)((nc + 255) / 256);
        init_out_kernel<<<ncb, 256, 0, stream>>>(w.hA, w.dinv, b1, w.hB);
        long long et = (long long)N_EDGES * 32;
        edge_scatter_kernel<<<(int)((et + 255) / 256), 256, 0, stream>>>(w.hA, w.hB, ei, w.dinv);
        fill_kernel<<<1, 256, 0, stream>>>(w.stats, 2 * HID, 0.0f);
        bn_stats_kernel<<<256, HID, 0, stream>>>(w.hB, w.stats);
        bn_finalize_kernel<<<1, HID, 0, stream>>>(w.stats, g1, be1);
        bn_gelu_kernel<<<ncb, 256, 0, stream>>>(w.hB, w.stats, w.hA, w.actbf);
    }
    // ---- layers 2,3 (K = 128, NKS = 4)
    run_layer(4, W2, b2, g2, be2, w, ei, stream);
    run_layer(4, W3, b3, g3, be3, w, ei, stream);

    // global mean pool
    fill_kernel<<<(NUM_GRAPHS * HID + NUM_GRAPHS + 255) / 256, 256, 0, stream>>>(
        w.pooled, (long long)NUM_GRAPHS * HID + NUM_GRAPHS, 0.0f);
    long long nc = (long long)N_NODES * HID;
    pool_accum_kernel<<<(int)((nc + 255) / 256), 256, 0, stream>>>(w.hA, batch, w.pooled, w.cnt);

    // MLP head
    head_kernel<<<NUM_GRAPHS, HALF_HID, 0, stream>>>(w.pooled, w.cnt, Wl1, bl1, Wl2, bl2, out);
}
